// SAGE_GraphConv_24592982737487
// MI455X (gfx1250) — compile-verified, matrix operand order checked by reading, not run on hardware
//
#include <hip/hip_runtime.h>
#include <hip/hip_bf16.h>

#define NNODES 10000
#define NEDGES 160000
#define DIM    512

typedef __attribute__((ext_vector_type(16))) __bf16 v16bf;
typedef __attribute__((ext_vector_type(8)))  __bf16 v8bf;
typedef __attribute__((ext_vector_type(8)))  float  v8f;
typedef __attribute__((ext_vector_type(2)))  unsigned int uv2;
typedef __attribute__((ext_vector_type(4)))  unsigned int uv4;
typedef __attribute__((ext_vector_type(4)))  int vi4;

typedef __attribute__((address_space(1))) vi4 gvi4;   // global <4 x i32>
typedef __attribute__((address_space(3))) vi4 lvi4;   // LDS    <4 x i32>

#if defined(__has_builtin)
#if __has_builtin(__builtin_amdgcn_global_load_async_to_lds_b128) && \
    __has_builtin(__builtin_amdgcn_s_wait_asynccnt)
#define HAVE_ASYNC_LDS 1
#endif
#endif

__device__ __forceinline__ unsigned short f2bf(float f) {
    unsigned int u = __builtin_bit_cast(unsigned int, f);
    u += 0x7fffu + ((u >> 16) & 1u);   // round-to-nearest-even
    return (unsigned short)(u >> 16);
}

// ---------------------------------------------------------------------------
// fp32 -> bf16 bulk convert (float4 -> 4 bf16 per thread)
// ---------------------------------------------------------------------------
__global__ __launch_bounds__(256) void f32_to_bf16_kernel(
    const float* __restrict__ in, unsigned short* __restrict__ out, int n4)
{
    int idx = blockIdx.x * 256 + threadIdx.x;
    if (idx >= n4) return;
    float4 v = ((const float4*)in)[idx];
    uv2 p;
    p.x = (unsigned)f2bf(v.x) | ((unsigned)f2bf(v.y) << 16);
    p.y = (unsigned)f2bf(v.z) | ((unsigned)f2bf(v.w) << 16);
    *(uv2*)(out + (size_t)idx * 4) = p;
}

// ---------------------------------------------------------------------------
// WMMA GEMM: C[m,n] = act( sum_k A[m,k]*W[n,k] (+bias[n]) (+C[m,n]) )
// A, W: bf16 row-major (A: [Mrows,512], W: [512,512] = [out,in]), C fp32.
// Block tile 128x128, BK=32, 256 threads = 8 waves, wave tile 32x64 (2x4 WMMA)
// Staging uses GLOBAL_LOAD_ASYNC_TO_LDS_B128 when available.
// ---------------------------------------------------------------------------
template <bool RELU, bool HAS_BIAS, bool ACCUM>
__global__ __launch_bounds__(256) void gemm_bf16_wmma(
    const unsigned short* __restrict__ Abf, const unsigned short* __restrict__ Wbf,
    const float* __restrict__ bias, float* __restrict__ C, int Mrows)
{
    constexpr int BM = 128, BN = 128, BK = 32;
    constexpr int LDT = BK + 8;                 // padded LDS stride (bf16 units)

    __shared__ unsigned short sA[BM * LDT];
    __shared__ unsigned short sB[BN * LDT];

    const int tid  = threadIdx.x;
    const int lane = tid & 31;
    const int wv   = tid >> 5;
    const int wm   = wv & 3;                    // wave M index 0..3 (32 rows each)
    const int wn   = wv >> 2;                   // wave N index 0..1 (64 cols each)
    const int lrow = lane & 15;
    const int koff = (lane >> 4) * 8;           // 0 or 8 (K sub-chunk)

    const int n0 = blockIdx.x * BN;             // 0..384
    const int m0 = blockIdx.y * BM;

    const v8f vzero = {0.f, 0.f, 0.f, 0.f, 0.f, 0.f, 0.f, 0.f};
    v8f acc[2][4];
    #pragma unroll
    for (int tm = 0; tm < 2; ++tm)
        #pragma unroll
        for (int tn = 0; tn < 4; ++tn) acc[tm][tn] = vzero;

    for (int kb = 0; kb < DIM; kb += BK) {
        // ---- stage A (128x32) and B (128x32) bf16 tiles into LDS ----
        #pragma unroll
        for (int i = 0; i < 2; ++i) {
            int idx = tid + i * 256;            // 0..511
            int row = idx >> 2;                 // 0..127
            int c   = (idx & 3) * 8;            // bf16 col: 0,8,16,24 (16B chunks)
            int gm  = m0 + row;
#if HAVE_ASYNC_LDS
            if (gm < Mrows) {
                __builtin_amdgcn_global_load_async_to_lds_b128(
                    (gvi4*)(Abf + (size_t)gm * DIM + kb + c),
                    (lvi4*)&sA[row * LDT + c], 0, 0);
            } else {
                uv4 z = {0u, 0u, 0u, 0u};
                *(uv4*)&sA[row * LDT + c] = z;
            }
            __builtin_amdgcn_global_load_async_to_lds_b128(
                (gvi4*)(Wbf + (size_t)(n0 + row) * DIM + kb + c),
                (lvi4*)&sB[row * LDT + c], 0, 0);
#else
            if (gm < Mrows) {
                *(uv4*)&sA[row * LDT + c] =
                    *(const uv4*)(Abf + (size_t)gm * DIM + kb + c);
            } else {
                uv4 z = {0u, 0u, 0u, 0u};
                *(uv4*)&sA[row * LDT + c] = z;
            }
            *(uv4*)&sB[row * LDT + c] =
                *(const uv4*)(Wbf + (size_t)(n0 + row) * DIM + kb + c);
            if (kb + BK < DIM && gm < Mrows)
                __builtin_prefetch(Abf + (size_t)gm * DIM + kb + BK + c, 0, 1);
#endif
        }
#if HAVE_ASYNC_LDS
        __builtin_amdgcn_s_wait_asynccnt(0);
#endif
        __syncthreads();

        // ---- fragments + WMMA ----
        v16bf af[2], bfr[4];
        #pragma unroll
        for (int tm = 0; tm < 2; ++tm) {
            int r = wm * 32 + tm * 16 + lrow;
            v8bf lo = *(const v8bf*)&sA[r * LDT + koff];
            v8bf hi = *(const v8bf*)&sA[r * LDT + 16 + koff];
            af[tm] = __builtin_shufflevector(lo, hi, 0,1,2,3,4,5,6,7,8,9,10,11,12,13,14,15);
        }
        #pragma unroll
        for (int tn = 0; tn < 4; ++tn) {
            int r = wn * 64 + tn * 16 + lrow;
            v8bf lo = *(const v8bf*)&sB[r * LDT + koff];
            v8bf hi = *(const v8bf*)&sB[r * LDT + 16 + koff];
            bfr[tn] = __builtin_shufflevector(lo, hi, 0,1,2,3,4,5,6,7,8,9,10,11,12,13,14,15);
        }
        #pragma unroll
        for (int tm = 0; tm < 2; ++tm)
            #pragma unroll
            for (int tn = 0; tn < 4; ++tn)
                acc[tm][tn] = __builtin_amdgcn_wmma_f32_16x16x32_bf16(
                    false, af[tm], false, bfr[tn], (short)0, acc[tm][tn], false, false);
        __syncthreads();
    }

    // ---- epilogue ----
    #pragma unroll
    for (int tm = 0; tm < 2; ++tm) {
        #pragma unroll
        for (int tn = 0; tn < 4; ++tn) {
            int nIdx  = n0 + wn * 64 + tn * 16 + lrow;
            int mBase = m0 + wm * 32 + tm * 16 + (lane >> 4) * 8;
            float bv = HAS_BIAS ? bias[nIdx] : 0.0f;
            #pragma unroll
            for (int i = 0; i < 8; ++i) {
                int m = mBase + i;
                if (m < Mrows) {
                    float v = acc[tm][tn][i] + bv;
                    if (ACCUM) v += C[(size_t)m * DIM + nIdx];
                    if (RELU)  v = fmaxf(v, 0.0f);
                    C[(size_t)m * DIM + nIdx] = v;
                }
            }
        }
    }
}

// ---------------------------------------------------------------------------
// Edge scatter: agg[dst] += h[src] (128 lanes per edge, float4 + f32 atomics)
// ---------------------------------------------------------------------------
__global__ __launch_bounds__(256) void scatter_feat(
    const float* __restrict__ h, const int* __restrict__ src,
    const int* __restrict__ dst, float* __restrict__ agg)
{
    int e = blockIdx.x * 2 + (threadIdx.x >> 7);
    if (e >= NEDGES) return;
    int c = (threadIdx.x & 127) * 4;
    int s = src[e], d = dst[e];
    float4 v = *(const float4*)&h[(size_t)s * DIM + c];
    float* a = &agg[(size_t)d * DIM + c];
    atomicAdd(a + 0, v.x);
    atomicAdd(a + 1, v.y);
    atomicAdd(a + 2, v.z);
    atomicAdd(a + 3, v.w);
}

__global__ __launch_bounds__(256) void scatter_cnt(
    const int* __restrict__ dst, float* __restrict__ cnt)
{
    int e = blockIdx.x * 256 + threadIdx.x;
    if (e < NEDGES) atomicAdd(&cnt[dst[e]], 1.0f);
}

// mean over incoming edges, fused with fp32 -> bf16 conversion
__global__ __launch_bounds__(256) void mean_bf16_kernel(
    const float* __restrict__ agg, const float* __restrict__ cnt,
    unsigned short* __restrict__ out)
{
    int idx = blockIdx.x * 256 + threadIdx.x;       // over N * (D/4)
    if (idx >= NNODES * (DIM / 4)) return;
    int row = idx / (DIM / 4);
    float s = 1.0f / fmaxf(cnt[row], 1.0f);
    float4 v = ((const float4*)agg)[idx];
    uv2 p;
    p.x = (unsigned)f2bf(v.x * s) | ((unsigned)f2bf(v.y * s) << 16);
    p.y = (unsigned)f2bf(v.z * s) | ((unsigned)f2bf(v.w * s) << 16);
    *(uv2*)(out + (size_t)idx * 4) = p;
}

// ---------------------------------------------------------------------------
// Row-wise ReLU + LayerNorm, fused with fp32 -> bf16 output (next layer's A)
// ---------------------------------------------------------------------------
__global__ __launch_bounds__(256) void relu_ln_bf16_kernel(
    const float* __restrict__ y, const float* __restrict__ g,
    const float* __restrict__ be, unsigned short* __restrict__ xout)
{
    __shared__ float red[256];
    int row = blockIdx.x, tid = threadIdx.x;
    const float2 v = ((const float2*)(y + (size_t)row * DIM))[tid];
    float v0 = fmaxf(v.x, 0.0f);
    float v1 = fmaxf(v.y, 0.0f);
    red[tid] = v0 + v1;
    __syncthreads();
    for (int o = 128; o > 0; o >>= 1) {
        if (tid < o) red[tid] += red[tid + o];
        __syncthreads();
    }
    float mu = red[0] * (1.0f / DIM);
    __syncthreads();
    float d0 = v0 - mu, d1 = v1 - mu;
    red[tid] = d0 * d0 + d1 * d1;
    __syncthreads();
    for (int o = 128; o > 0; o >>= 1) {
        if (tid < o) red[tid] += red[tid + o];
        __syncthreads();
    }
    float rs = rsqrtf(red[0] * (1.0f / DIM) + 1e-5f);
    float o0 = d0 * rs * g[2 * tid]     + be[2 * tid];
    float o1 = d1 * rs * g[2 * tid + 1] + be[2 * tid + 1];
    unsigned int pk = (unsigned)f2bf(o0) | ((unsigned)f2bf(o1) << 16);
    *(unsigned int*)(xout + (size_t)row * DIM + 2 * tid) = pk;
}

// ---------------------------------------------------------------------------
extern "C" void kernel_launch(void* const* d_in, const int* in_sizes, int n_in,
                              void* d_out, int out_size, void* d_ws, size_t ws_size,
                              hipStream_t stream)
{
    const float* x   = (const float*)d_in[0];
    const int*   ei  = (const int*)d_in[1];
    const int*   src = ei;
    const int*   dst = ei + NEDGES;

    const float *Wp[3], *bp[3], *Wl[3], *bl[3], *Wr[3], *g[2], *be[2];
    int i = 2;
    for (int l = 0; l < 3; ++l) {
        Wp[l] = (const float*)d_in[i++];
        bp[l] = (const float*)d_in[i++];
        Wl[l] = (const float*)d_in[i++];
        bl[l] = (const float*)d_in[i++];
        Wr[l] = (const float*)d_in[i++];
        if (l < 2) {
            g[l]  = (const float*)d_in[i++];
            be[l] = (const float*)d_in[i++];
        }
    }

    const size_t ND = (size_t)NNODES * DIM;     // 5,120,000
    const size_t WD = (size_t)DIM * DIM;        // 262,144
    float*          hY    = (float*)d_ws;                 // ND fp32 (h, then y)
    float*          agg   = hY + ND;                      // ND fp32
    float*          cnt   = agg + ND;                     // NNODES fp32
    unsigned short* xbf   = (unsigned short*)(cnt + NNODES); // ND bf16
    unsigned short* aggbf = xbf + ND;                     // ND bf16
    unsigned short* wbf   = aggbf + ND;                   // 9 * WD bf16
    float*          out   = (float*)d_out;

    dim3 blk(256);
    dim3 gemmGrid(DIM / 128, (NNODES + 127) / 128);       // (4, 79)

    // ---- one-time (per call) bf16 conversions: input x + 9 weight matrices ----
    f32_to_bf16_kernel<<<(int)(ND / 4 + 255) / 256, blk, 0, stream>>>(x, xbf, (int)(ND / 4));
    const float* wsrc[9] = {Wp[0], Wl[0], Wr[0], Wp[1], Wl[1], Wr[1], Wp[2], Wl[2], Wr[2]};
    for (int w = 0; w < 9; ++w)
        f32_to_bf16_kernel<<<(int)(WD / 4 + 255) / 256, blk, 0, stream>>>(
            wsrc[w], wbf + (size_t)w * WD, (int)(WD / 4));

    auto wp = [&](int l) { return wbf + (size_t)(3 * l + 0) * WD; };
    auto wl = [&](int l) { return wbf + (size_t)(3 * l + 1) * WD; };
    auto wr = [&](int l) { return wbf + (size_t)(3 * l + 2) * WD; };

    for (int l = 0; l < 3; ++l) {
        float* y = (l == 2) ? out : hY;   // y reuses h buffer (h dead after scatter)
        // h = relu(x @ Wp^T + bp)
        gemm_bf16_wmma<true, true, false><<<gemmGrid, blk, 0, stream>>>(
            xbf, wp(l), bp[l], hY, NNODES);
        (void)hipMemsetAsync(agg, 0, ND * sizeof(float), stream);
        (void)hipMemsetAsync(cnt, 0, NNODES * sizeof(float), stream);
        scatter_feat<<<NEDGES / 2, blk, 0, stream>>>(hY, src, dst, agg);
        scatter_cnt<<<(NEDGES + 255) / 256, blk, 0, stream>>>(dst, cnt);
        mean_bf16_kernel<<<(int)(ND / 4 + 255) / 256, blk, 0, stream>>>(agg, cnt, aggbf);
        // y = agg @ Wl^T + bl ; y += x @ Wr^T
        gemm_bf16_wmma<false, true, false><<<gemmGrid, blk, 0, stream>>>(
            aggbf, wl(l), bl[l], y, NNODES);
        gemm_bf16_wmma<false, false, true><<<gemmGrid, blk, 0, stream>>>(
            xbf, wr(l), nullptr, y, NNODES);
        if (l < 2)
            relu_ln_bf16_kernel<<<NNODES, blk, 0, stream>>>(y, g[l], be[l], xbf);
    }
}
